// DownsampleAttention_51943334477848
// MI455X (gfx1250) — matrix operand order repaired and checked by reading, not compiled, not run on hardware
//
#include <hip/hip_runtime.h>

typedef __attribute__((ext_vector_type(16))) _Float16 v16h;
typedef __attribute__((ext_vector_type(8)))  _Float16 v8h;
typedef __attribute__((ext_vector_type(8)))  float    v8f;
typedef __attribute__((ext_vector_type(4)))  float    v4f;

#define WMMA_F16(a, b, c) \
  __builtin_amdgcn_wmma_f32_16x16x32_f16(false, (a), false, (b), (short)0, (c), false, false)

static constexpr int NB = 2;      // batch
static constexpr int S  = 4096;   // input seq
static constexpr int D  = 512;    // model dim
static constexpr int L  = 2048;   // downsampled seq (S/2)
static constexpr int H  = 8;      // heads
static constexpr int HD = 64;     // head dim
static constexpr int DP = 1024;   // D * P (projection input dim)

// ---------------------------------------------------------------------------
// Fragment layouts (CDNA5 ISA 7.12.2), wave32:
//  A (16x32 f16): lane holds row M = lane%16; K-values are two contiguous
//    runs of 8:  [hf*16, hf*16+8) and [hf*16+8, hf*16+16)   (hf = lane>=16; this file's order)
//  B (32x16 f16): lane holds col N = lane%16; K-values (this file's order, as A) are one contiguous
//    run of 16:  [hf*16, hf*16+16)
//  C (16x16 f32): VGPR r -> row r + hf*8, col = lane%16
// So: B fragments are a single 32B load if source is K-contiguous per column;
//     A fragments are two 16B loads if source is K-contiguous per row.
// ---------------------------------------------------------------------------

// ---------------------------------------------------------------------------
// Kernel 0: convert weights to f16, transposed to N-major (Wt[col][k]) so
// every B fragment in the GEMMs is one contiguous v16h load.
// ---------------------------------------------------------------------------
__global__ __launch_bounds__(256)
void wconv_kernel(const float* __restrict__ Wq, const float* __restrict__ Wk,
                  const float* __restrict__ Wv, const float* __restrict__ Wo,
                  _Float16* __restrict__ Wqt, _Float16* __restrict__ Wkt,
                  _Float16* __restrict__ Wvt, _Float16* __restrict__ Wot)
{
  const int mat = blockIdx.y;
  const float* W; _Float16* Wt; int K;
  if      (mat == 0) { W = Wq; Wt = Wqt; K = DP; }
  else if (mat == 1) { W = Wk; Wt = Wkt; K = DP; }
  else if (mat == 2) { W = Wv; Wt = Wvt; K = DP; }
  else               { W = Wo; Wt = Wot; K = D;  }
  const int idx = blockIdx.x * 256 + threadIdx.x;   // over D*K, k fastest
  if (idx < D * K) {
    const int c = idx / K, k = idx % K;
    Wt[(size_t)c * K + k] = (_Float16)W[(size_t)k * D + c];
  }
}

// ---------------------------------------------------------------------------
// Kernel 1: fused downsample + QKV projection.
// One wave per 16x16 tile; K = 1024 in steps of 32.
// xd[n,l,f] = x[n, 2l + (f&1), f>>1]: A pairs (f,f+1) with f even come from
// rows (2l, 2l+1) at feature f>>1 -> A fragment = 4x float4 loads, converted.
// V is written TRANSPOSED (vt[n][d][kv]) so attention's P*V B-fragments are
// contiguous v16h loads.
// ---------------------------------------------------------------------------
__global__ __launch_bounds__(32)
void qkv_kernel(const float* __restrict__ x,
                const _Float16* __restrict__ Wqt, const float* __restrict__ bq,
                const _Float16* __restrict__ Wkt, const float* __restrict__ bk,
                const _Float16* __restrict__ Wvt, const float* __restrict__ bv,
                _Float16* __restrict__ qo, _Float16* __restrict__ ko,
                _Float16* __restrict__ vt)
{
  const int lane = threadIdx.x;
  const int hf = lane >> 4, m = lane & 15;
  const int rt = blockIdx.x;
  const int yy = blockIdx.y;
  const int mat = yy >> 5, ct = yy & 31;

  const _Float16* Wt; const float* bias;
  if      (mat == 0) { Wt = Wqt; bias = bq; }
  else if (mat == 1) { Wt = Wkt; bias = bk; }
  else               { Wt = Wvt; bias = bv; }

  const int row = rt * 16 + m;            // 0..4095 over n*L
  const int n = row >> 11, l = row & 2047;
  const float* xr0 = x + (size_t)(n * S + 2 * l) * D;   // p = 0 source row
  const float* xr1 = xr0 + D;                           // p = 1 source row
  const int col = ct * 16 + m;
  const _Float16* wrow = Wt + (size_t)col * DP;

  v8f acc = {};
#pragma unroll 2
  for (int kk = 0; kk < DP; kk += 32) {
    if (kk + 32 < DP)
      __builtin_prefetch(wrow + kk + 32, 0, 1);
    // A: two contiguous float4 runs per source row (downsample is free).
    const int f2 = (kk >> 1) + hf * 8;
    const v4f c00 = *(const v4f*)(xr0 + f2);
    const v4f c01 = *(const v4f*)(xr0 + f2 + 4);
    const v4f c10 = *(const v4f*)(xr1 + f2);
    const v4f c11 = *(const v4f*)(xr1 + f2 + 4);
    v16h a;
#pragma unroll
    for (int j = 0; j < 4; ++j) {
      a[2 * j]     = (_Float16)c00[j];
      a[2 * j + 1] = (_Float16)c10[j];
      a[8 + 2 * j]     = (_Float16)c01[j];
      a[8 + 2 * j + 1] = (_Float16)c11[j];
    }
    // B: one contiguous 32B load (K-contiguous, lane-half selects K block).
    const v16h b = *(const v16h*)(wrow + kk + hf * 16);
    acc = WMMA_F16(a, b, acc);
  }
  const float bb = bias[col];
#pragma unroll
  for (int r = 0; r < 8; ++r) {
    const int rg = rt * 16 + r + hf * 8;
    const float val = acc[r] + bb;
    if (mat == 0) {
      qo[(size_t)rg * D + col] = (_Float16)val;
    } else if (mat == 1) {
      ko[(size_t)rg * D + col] = (_Float16)val;
    } else {
      const int nn = rg >> 11, ll = rg & 2047;          // V transposed
      vt[((size_t)nn * D + col) * L + ll] = (_Float16)val;
    }
  }
}

// ---------------------------------------------------------------------------
// Kernel 2: causal flash-attention, one wave per (batch, head, 16-row q block).
// S = Q(16x64) * K^T via 4 WMMAs per 32-wide kv block (K fragments are single
// v16h loads), online softmax with width-16 shfl row reductions, P staged
// through LDS (C-layout -> A-layout via 2x ds b128), O += P*V via 4 WMMAs
// with V fragments as single v16h loads from the transposed V buffer.
// ---------------------------------------------------------------------------
__global__ __launch_bounds__(32)
void attn_kernel(const _Float16* __restrict__ qb_, const _Float16* __restrict__ kb_,
                 const _Float16* __restrict__ vt_, _Float16* __restrict__ ao)
{
  __shared__ _Float16 lds_p[16 * 32];

  const int lane = threadIdx.x, hf = lane >> 4, m = lane & 15;
  const int qblk = blockIdx.x, h = blockIdx.y, n = blockIdx.z;
  const float scale = 0.04419417382415922f;   // 512^-0.5

  const size_t base = (size_t)n * L * D + (size_t)h * HD;

  // Q fragments: row = qblk*16 + m, two v8h runs per 32-K fragment.
  const _Float16* qp = qb_ + base + (size_t)(qblk * 16 + m) * D;
  const v8h q0lo = *(const v8h*)(qp + hf * 16);
  const v8h q0hi = *(const v8h*)(qp + 8 + hf * 16);
  const v8h q1lo = *(const v8h*)(qp + 32 + hf * 16);
  const v8h q1hi = *(const v8h*)(qp + 40 + hf * 16);
  v16h aq0, aq1;
#pragma unroll
  for (int e = 0; e < 8; ++e) {
    aq0[e] = q0lo[e]; aq0[8 + e] = q0hi[e];
    aq1[e] = q1lo[e]; aq1[8 + e] = q1hi[e];
  }

  float mi[8], li[8];
  v8f o0 = {}, o1 = {}, o2 = {}, o3 = {};
#pragma unroll
  for (int r = 0; r < 8; ++r) { mi[r] = -1e30f; li[r] = 0.0f; }

  const int jb_end = (qblk * 16 + 15) >> 5;   // last kv block (inclusive, causal)
  for (int jb = 0; jb <= jb_end; ++jb) {
    const int kv0 = jb * 32;

    // K^T fragments: lane's column = kv row; K-contiguous -> single v16h loads.
    const _Float16* kp0 = kb_ + base + (size_t)(kv0 + m) * D;
    const _Float16* kp1 = kb_ + base + (size_t)(kv0 + 16 + m) * D;
    const v16h b0lo = *(const v16h*)(kp0 + hf * 16);
    const v16h b0hi = *(const v16h*)(kp0 + 32 + hf * 16);
    const v16h b1lo = *(const v16h*)(kp1 + hf * 16);
    const v16h b1hi = *(const v16h*)(kp1 + 32 + hf * 16);

    v8f s0 = {}, s1 = {};
    s0 = WMMA_F16(aq0, b0lo, s0);
    s0 = WMMA_F16(aq1, b0hi, s0);
    s1 = WMMA_F16(aq0, b1lo, s1);
    s1 = WMMA_F16(aq1, b1hi, s1);

    // Scale, causal mask, online softmax update.
    const int c0 = kv0 + m, c1 = kv0 + 16 + m;
#pragma unroll
    for (int r = 0; r < 8; ++r) {
      const int rowa = qblk * 16 + r + hf * 8;
      float v0 = s0[r] * scale; if (c0 > rowa) v0 = -1e30f;
      float v1 = s1[r] * scale; if (c1 > rowa) v1 = -1e30f;
      float mx = fmaxf(v0, v1);
      for (int off = 1; off < 16; off <<= 1)
        mx = fmaxf(mx, __shfl_xor(mx, off, 16));
      const float mnew = fmaxf(mi[r], mx);
      const float alpha = __expf(mi[r] - mnew);
      v0 = __expf(v0 - mnew);
      v1 = __expf(v1 - mnew);
      float rs = v0 + v1;
      for (int off = 1; off < 16; off <<= 1)
        rs += __shfl_xor(rs, off, 16);
      li[r] = li[r] * alpha + rs;
      mi[r] = mnew;
      o0[r] *= alpha; o1[r] *= alpha; o2[r] *= alpha; o3[r] *= alpha;
      // P to LDS, row-major [row][kvcol] (C-layout scatter)
      lds_p[(r + hf * 8) * 32 + m]      = (_Float16)v0;
      lds_p[(r + hf * 8) * 32 + 16 + m] = (_Float16)v1;
    }
    __syncthreads();

    // Reload P in A-fragment layout: two contiguous v8h runs -> ds b128.
    const v8h plo = *(const v8h*)(&lds_p[m * 32 + hf * 16]);
    const v8h phi = *(const v8h*)(&lds_p[m * 32 + 8 + hf * 16]);
    v16h ap;
#pragma unroll
    for (int e = 0; e < 8; ++e) { ap[e] = plo[e]; ap[8 + e] = phi[e]; }
    __syncthreads();

    // O += P * V: V fragment per column tile = single v16h load (kv contiguous
    // in the transposed V buffer).
    const _Float16* vp = vt_ + ((size_t)n * D + h * HD) * L + kv0 + hf * 16;
#pragma unroll
    for (int t = 0; t < 4; ++t) {
      const v16h bvf = *(const v16h*)(vp + (size_t)(t * 16 + m) * L);
      if      (t == 0) o0 = WMMA_F16(ap, bvf, o0);
      else if (t == 1) o1 = WMMA_F16(ap, bvf, o1);
      else if (t == 2) o2 = WMMA_F16(ap, bvf, o2);
      else             o3 = WMMA_F16(ap, bvf, o3);
    }
  }

  // Normalize and store attention output (f16, row-major).
#pragma unroll
  for (int r = 0; r < 8; ++r) {
    const float inv = 1.0f / li[r];
    const int rg = qblk * 16 + r + hf * 8;
    _Float16* op = ao + base + (size_t)rg * D;
    op[0 * 16 + m] = (_Float16)(o0[r] * inv);
    op[1 * 16 + m] = (_Float16)(o1[r] * inv);
    op[2 * 16 + m] = (_Float16)(o2[r] * inv);
    op[3 * 16 + m] = (_Float16)(o3[r] * inv);
  }
}

// ---------------------------------------------------------------------------
// Kernel 3: output projection ao(4096x512 f16) @ Wo + bo -> f32 out.
// A: two v8h loads; B: one v16h load from transposed f16 Wo.
// ---------------------------------------------------------------------------
__global__ __launch_bounds__(32)
void proj_kernel(const _Float16* __restrict__ ao, const _Float16* __restrict__ Wot,
                 const float* __restrict__ bo, float* __restrict__ out)
{
  const int lane = threadIdx.x, hf = lane >> 4, m = lane & 15;
  const int rt = blockIdx.x, ct = blockIdx.y;
  const int row = rt * 16 + m;
  const int col = ct * 16 + m;
  const _Float16* ar = ao + (size_t)row * D;
  const _Float16* wrow = Wot + (size_t)col * D;

  v8f acc = {};
#pragma unroll 2
  for (int kk = 0; kk < D; kk += 32) {
    if (kk + 32 < D)
      __builtin_prefetch(wrow + kk + 32, 0, 1);
    const v8h alo = *(const v8h*)(ar + kk + hf * 16);
    const v8h ahi = *(const v8h*)(ar + kk + 8 + hf * 16);
    v16h a;
#pragma unroll
    for (int e = 0; e < 8; ++e) { a[e] = alo[e]; a[8 + e] = ahi[e]; }
    const v16h b = *(const v16h*)(wrow + kk + hf * 16);
    acc = WMMA_F16(a, b, acc);
  }
  const float bb = bo[col];
#pragma unroll
  for (int r = 0; r < 8; ++r) {
    const int rg = rt * 16 + r + hf * 8;
    out[(size_t)rg * D + col] = acc[r] + bb;
  }
}

// ---------------------------------------------------------------------------
extern "C" void kernel_launch(void* const* d_in, const int* in_sizes, int n_in,
                              void* d_out, int out_size, void* d_ws, size_t ws_size,
                              hipStream_t stream) {
  (void)in_sizes; (void)n_in; (void)out_size; (void)ws_size;
  const float* x  = (const float*)d_in[0];
  const float* Wq = (const float*)d_in[1];
  const float* bq = (const float*)d_in[2];
  const float* Wk = (const float*)d_in[3];
  const float* bk = (const float*)d_in[4];
  const float* Wv = (const float*)d_in[5];
  const float* bv = (const float*)d_in[6];
  const float* Wo = (const float*)d_in[7];
  const float* bo = (const float*)d_in[8];
  float* out = (float*)d_out;

  const size_t nld = (size_t)NB * L * D;        // 2,097,152 elements
  _Float16* qws = (_Float16*)d_ws;              // activations: 4 x 4 MB
  _Float16* kws = qws + nld;
  _Float16* vtw = kws + nld;                    // V, transposed [n][d][kv]
  _Float16* aws = vtw + nld;
  _Float16* wqt = aws + nld;                    // f16 transposed weights
  _Float16* wkt = wqt + (size_t)D * DP;
  _Float16* wvt = wkt + (size_t)D * DP;
  _Float16* wot = wvt + (size_t)D * DP;

  dim3 g0((D * DP + 255) / 256, 4);             // weight convert/transpose
  wconv_kernel<<<g0, 256, 0, stream>>>(Wq, Wk, Wv, Wo, wqt, wkt, wvt, wot);

  dim3 g1(256, 96);        // (n*L)/16 row tiles x (3 matrices * 32 col tiles)
  qkv_kernel<<<g1, 32, 0, stream>>>(x, wqt, bq, wkt, bk, wvt, bv, qws, kws, vtw);

  dim3 g2(128, H, NB);     // q blocks x heads x batch
  attn_kernel<<<g2, 32, 0, stream>>>(qws, kws, vtw, aws);

  dim3 g3(256, 32);        // (n*L)/16 x D/16
  proj_kernel<<<g3, 32, 0, stream>>>(aws, wot, bo, out);
}
